// MambaDSN_28011776704904
// MI455X (gfx1250) — matrix operand order repaired and checked by reading, not compiled, not run on hardware
//
#include <hip/hip_runtime.h>
#include <hip/hip_bf16.h>
#include <math.h>

// ---------------------------------------------------------------------------
// Types for CDNA5 WMMA (wave32): v_wmma_f32_16x16x32_bf16
// ---------------------------------------------------------------------------
typedef __bf16 bf16_t;
typedef __bf16 v16bf __attribute__((ext_vector_type(16)));
typedef __bf16 v8bf  __attribute__((ext_vector_type(8)));
typedef __bf16 v4bf  __attribute__((ext_vector_type(4)));
typedef float  v8f   __attribute__((ext_vector_type(8)));

#define LDSTR 40  // padded LDS row stride in bf16 elems (80B rows: 16B aligned, conflict-free gathers)

enum { EPI_NONE = 0, EPI_BIAS = 1, EPI_BIAS_SOFTPLUS = 2 };

__device__ __forceinline__ float softplus_f(float x) {
  return (x > 20.f) ? x : log1pf(__expf(x));
}
__device__ __forceinline__ float silu_f(float x) {
  return x * (1.f / (1.f + __expf(-x)));
}

// ---------------------------------------------------------------------------
// C[M,N] = A[M,K](lda=LDA) @ W[N,K]^T  (+bias / +softplus)
// Compile-time K/N/LDA so all edge guards fold to straight-line code.
// block = 256 threads = 8 waves (4 M-waves x 2 N-waves), tile = 64(M) x 32(N).
// Double-buffered LDS, one barrier per 32-deep K step, one WMMA per wave/step.
// ---------------------------------------------------------------------------
template <int EPI, int K, int N, int LDA>
__global__ __launch_bounds__(256)
void gemm_bf16_wmma(const float* __restrict__ A,
                    const float* __restrict__ W,     // (N,K) row-major
                    const float* __restrict__ bias,  // (N) or nullptr
                    float* __restrict__ C) {
  constexpr int KTILES = (K + 31) / 32;
  __shared__ __attribute__((aligned(16))) bf16_t As[2][64 * LDSTR];
  __shared__ __attribute__((aligned(16))) bf16_t Bs[2][32 * LDSTR];

  const int tid   = threadIdx.x;
  const int lane  = tid & 31;
  const int wave  = tid >> 5;
  const int waveM = wave & 3;   // 0..3
  const int waveN = wave >> 2;  // 0..1

  const int m0 = blockIdx.x * 64;
  const int n0 = blockIdx.y * 32;

  // staging coordinates
  const int ar  = tid >> 2;         // 0..63  (A row in tile)
  const int akk = (tid & 3) * 8;    // 0,8,16,24
  const int bn  = tid >> 3;         // 0..31  (W row == B column)
  const int bkk = (tid & 7) * 4;    // 0..28

  const long argrow = (long)(m0 + ar) * LDA;
  const int  bgn    = n0 + bn;
  const long bwrow  = (long)bgn * K;

  // ---- stagers: straight-line (K multiple of 8; OOB groups become zero-fill)
  auto stageA = [&](int k0, int buf) {
    v8bf p = {};
    if (k0 + akk + 8 <= K) {  // uniform per thread; no partial groups (K%8==0)
      const float4 f0 = *(const float4*)&A[argrow + k0 + akk];
      const float4 f1 = *(const float4*)&A[argrow + k0 + akk + 4];
      p[0] = (bf16_t)f0.x; p[1] = (bf16_t)f0.y; p[2] = (bf16_t)f0.z; p[3] = (bf16_t)f0.w;
      p[4] = (bf16_t)f1.x; p[5] = (bf16_t)f1.y; p[6] = (bf16_t)f1.z; p[7] = (bf16_t)f1.w;
    }
    *(v8bf*)&As[buf][ar * LDSTR + akk] = p;   // one ds_store_b128
  };
  auto stageB = [&](int k0, int buf) {
    v4bf p = {};
    if (bgn < N && k0 + bkk + 4 <= K) {
      const float4 f = *(const float4*)&W[bwrow + k0 + bkk];
      p[0] = (bf16_t)f.x; p[1] = (bf16_t)f.y; p[2] = (bf16_t)f.z; p[3] = (bf16_t)f.w;
    }
    *(v4bf*)&Bs[buf][bn * LDSTR + bkk] = p;   // one ds_store_b64
  };

  v8f acc = {};

  stageA(0, 0);
  stageB(0, 0);
  __syncthreads();

  // fragment addressing (ISA 7.12.2 lane layouts)
  const int arow  = waveM * 16 + (lane & 15);
  const int kbase = (lane >> 4) * 8;
  const int bcol  = waveN * 16 + (lane & 15);
  const int koff  = (lane >> 4) * 16;

  for (int t = 0; t < KTILES; ++t) {
    const int cur = t & 1;
    if (t + 1 < KTILES) {
      stageA((t + 1) * 32, cur ^ 1);
      stageB((t + 1) * 32, cur ^ 1);
      if (t + 2 < KTILES) __builtin_prefetch(&A[argrow + (t + 2) * 32 + akk], 0, 1);
    }

    union { v16bf v; v8bf h[2]; } fa, fb;
    fa.h[0] = *(const v8bf*)&As[cur][arow * LDSTR + kbase];
    fa.h[1] = *(const v8bf*)&As[cur][arow * LDSTR + kbase + 16];
    fb.h[0] = *(const v8bf*)&Bs[cur][bcol * LDSTR + kbase];
    fb.h[1] = *(const v8bf*)&Bs[cur][bcol * LDSTR + kbase + 16];

    acc = __builtin_amdgcn_wmma_f32_16x16x32_bf16(
        /*neg_a=*/false, fa.v, /*neg_b=*/false, fb.v,
        /*c_mod=*/(short)0, acc, /*reuse_a=*/false, /*reuse_b=*/false);
    __syncthreads();   // single barrier per step (2 LDS buffers make this safe)
  }

  // ---- epilogue + store: D layout: VGPR r -> M = r + (lane>>4)*8, N = lane&15
  const int n = n0 + waveN * 16 + (lane & 15);
  if (n < N) {
    float bv = 0.f;
    if (EPI != EPI_NONE && bias) bv = bias[n];
    const int mrow0 = m0 + waveM * 16 + (lane >> 4) * 8;
#pragma unroll
    for (int r = 0; r < 8; ++r) {
      float v = acc[r];
      if (EPI == EPI_BIAS)          v += bv;
      if (EPI == EPI_BIAS_SOFTPLUS) v = softplus_f(v + bv);
      C[(long)(mrow0 + r) * N + n] = v;
    }
  }
}

// ---------------------------------------------------------------------------
// Causal depthwise conv (DCONV=4) over time + bias + SiLU.
// x is the first half of xz (B,L,1024); output u (B,L,512).
// ---------------------------------------------------------------------------
__global__ __launch_bounds__(256)
void conv_silu_kernel(const float* __restrict__ xz,
                      const float* __restrict__ conv_w,
                      const float* __restrict__ conv_b,
                      float* __restrict__ u, int total, int L) {
  int idx = blockIdx.x * blockDim.x + threadIdx.x;  // over B*L*512
  if (idx >= total) return;
  int d   = idx & 511;
  int row = idx >> 9;        // b*L + l
  int l   = row & (L - 1);   // L is a power of two
  float acc = conv_b[d];
#pragma unroll
  for (int j = 0; j < 4; ++j) {
    int lj = l - 3 + j;
    if (lj >= 0) acc += conv_w[d * 4 + j] * xz[(long)(row - 3 + j) * 1024 + d];
  }
  u[(long)row * 512 + d] = silu_f(acc);
}

// ---------------------------------------------------------------------------
// Selective scan: one lane per (b, d, n); 16-lane shfl_xor tree for h . C.
// Fuses +u*D and the silu(z) gate. 32768 lanes = 1024 waves.
// ---------------------------------------------------------------------------
__global__ __launch_bounds__(256)
void scan_kernel(const float* __restrict__ dt, const float* __restrict__ u,
                 const float* __restrict__ xdbl, const float* __restrict__ xz,
                 const float* __restrict__ A_log, const float* __restrict__ D,
                 float* __restrict__ y, int L) {
  int g = blockIdx.x * blockDim.x + threadIdx.x;  // 0 .. B*512*16-1
  int n = g & 15;
  int d = (g >> 4) & 511;
  int b = g >> 13;
  float Adn = -__expf(A_log[d * 16 + n]);
  float Dd  = D[d];
  float s   = 0.f;
  long rowBase = (long)b * L;
  for (int l = 0; l < L; ++l) {
    long row = rowBase + l;
    float dtv = dt[row * 512 + d];
    float uv  = u[row * 512 + d];
    float Bt  = xdbl[row * 48 + 16 + n];
    float Ct  = xdbl[row * 48 + 32 + n];
    s = __expf(dtv * Adn) * s + (dtv * uv) * Bt;
    float yv = s * Ct;
    yv += __shfl_xor(yv, 1, 32);
    yv += __shfl_xor(yv, 2, 32);
    yv += __shfl_xor(yv, 4, 32);
    yv += __shfl_xor(yv, 8, 32);
    if (n == 0) {
      float zv = xz[row * 1024 + 512 + d];
      y[row * 512 + d] = (yv + uv * Dd) * silu_f(zv);
    }
  }
}

// ---------------------------------------------------------------------------
// fc_out: one wave per row, K=256 dot + sigmoid.
// ---------------------------------------------------------------------------
__global__ __launch_bounds__(256)
void fcout_kernel(const float* __restrict__ h, const float* __restrict__ Wout,
                  const float* __restrict__ bout, float* __restrict__ out, int M) {
  int wave = (blockIdx.x * blockDim.x + threadIdx.x) >> 5;
  int lane = threadIdx.x & 31;
  if (wave >= M) return;
  float s = 0.f;
#pragma unroll
  for (int i = 0; i < 8; ++i) {
    int k = lane + i * 32;
    s += h[(long)wave * 256 + k] * Wout[k];
  }
  for (int off = 16; off > 0; off >>= 1) s += __shfl_down(s, off, 32);
  if (lane == 0) out[wave] = 1.f / (1.f + __expf(-(s + bout[0])));
}

// ---------------------------------------------------------------------------
extern "C" void kernel_launch(void* const* d_in, const int* in_sizes, int n_in,
                              void* d_out, int out_size, void* d_ws, size_t ws_size,
                              hipStream_t stream) {
  const float* x          = (const float*)d_in[0];
  const float* W_in       = (const float*)d_in[1];
  const float* b_in       = (const float*)d_in[2];
  const float* in_proj_w  = (const float*)d_in[3];
  const float* conv_w     = (const float*)d_in[4];
  const float* conv_b     = (const float*)d_in[5];
  const float* x_proj_w   = (const float*)d_in[6];
  const float* dt_proj_w  = (const float*)d_in[7];
  const float* dt_proj_b  = (const float*)d_in[8];
  const float* A_log      = (const float*)d_in[9];
  const float* Dvec       = (const float*)d_in[10];
  const float* out_proj_w = (const float*)d_in[11];
  const float* W_out      = (const float*)d_in[12];
  const float* b_out      = (const float*)d_in[13];
  float* out = (float*)d_out;

  constexpr int Bc = 4, Lc = 8192, DINc = 192, Hc = 256, DI = 512, Nc = 16, Rc = 16;
  const int M = Bc * Lc;  // 32768

  // workspace layout (floats)
  float* ws = (float*)d_ws;
  size_t o = 0;
  float* hbuf  = ws + o; o += (size_t)M * Hc;             // fc_in result; reused for out_proj result
  float* xzbuf = ws + o; o += (size_t)M * 2 * DI;         // in_proj result (x | z)
  float* ubuf  = ws + o; o += (size_t)M * DI;             // conv+silu
  float* xdbl  = ws + o; o += (size_t)M * (Rc + 2 * Nc);  // [dt_r | B | C]
  float* dtbuf = ws + o; o += (size_t)M * DI;             // softplus(dt)
  float* ybuf  = ws + o; o += (size_t)M * DI;             // gated scan output

  dim3 blk(256);
  auto gg = [&](int n) { return dim3(M / 64, (n + 31) / 32); };

  // 1. fc_in: h = x @ W_in^T + b_in                       (M,192)->(M,256)
  gemm_bf16_wmma<EPI_BIAS, 192, 256, 192>
      <<<gg(Hc), blk, 0, stream>>>(x, W_in, b_in, hbuf);
  // 2. in_proj: xz = h @ in_proj_w^T                      (M,256)->(M,1024)
  gemm_bf16_wmma<EPI_NONE, 256, 1024, 256>
      <<<gg(2 * DI), blk, 0, stream>>>(hbuf, in_proj_w, nullptr, xzbuf);
  // 3. causal depthwise conv + SiLU -> u                  (M,512)
  conv_silu_kernel<<<(M * DI) / 256, blk, 0, stream>>>(xzbuf, conv_w, conv_b, ubuf, M * DI, Lc);
  // 4. x_proj: x_dbl = u @ x_proj_w^T                     (M,512)->(M,48)
  gemm_bf16_wmma<EPI_NONE, 512, 48, 512>
      <<<gg(Rc + 2 * Nc), blk, 0, stream>>>(ubuf, x_proj_w, nullptr, xdbl);
  // 5. dt = softplus(x_dbl[:, :16] @ dt_proj_w^T + b)     (M,16)->(M,512); lda=48
  gemm_bf16_wmma<EPI_BIAS_SOFTPLUS, 16, 512, 48>
      <<<gg(DI), blk, 0, stream>>>(xdbl, dt_proj_w, dt_proj_b, dtbuf);
  // 6. selective scan (+u*D, *silu(z)) -> ybuf
  scan_kernel<<<(Bc * DI * Nc) / 256, blk, 0, stream>>>(dtbuf, ubuf, xdbl, xzbuf, A_log, Dvec, ybuf, Lc);
  // 7. out_proj: h2 = y @ out_proj_w^T (reuse hbuf)       (M,512)->(M,256)
  gemm_bf16_wmma<EPI_NONE, 512, 256, 512>
      <<<gg(Hc), blk, 0, stream>>>(ybuf, out_proj_w, nullptr, hbuf);
  // 8. fc_out: sigmoid(h2 @ W_out^T + b_out)              (M,256)->(M,1)
  fcout_kernel<<<(M * 32) / 256, blk, 0, stream>>>(hbuf, W_out, b_out, out, M);
}
